// ArflowSparseMoeBlock_35175782154722
// MI455X (gfx1250) — matrix operand order repaired and checked by reading, not compiled, not run on hardware
//
#include <hip/hip_runtime.h>
#include <hip/hip_bf16.h>
#include <cstdint>
#include <cstddef>

#define IN_F   12336
#define INT_F  1024
#define OUT_F  96
#define NE     8
#define NT     1024   // tokens = B*S = 2*512

typedef __attribute__((ext_vector_type(16))) __bf16 v16bf;
typedef __attribute__((ext_vector_type(8)))  __bf16 v8bf;
typedef __attribute__((ext_vector_type(8)))  float  v8f;
typedef __attribute__((ext_vector_type(4)))  int    v4i;

// -------- async global->LDS (gfx1250) with compile-safe fallback -----------
#if defined(__has_builtin)
#if __has_builtin(__builtin_amdgcn_global_load_async_to_lds_b128)
#define HAVE_ASYNC_LDS 1
#endif
#endif
#ifndef HAVE_ASYNC_LDS
#define HAVE_ASYNC_LDS 0
#endif

#if HAVE_ASYNC_LDS
typedef __attribute__((address_space(1))) v4i gv4i;  // global 16B payload
typedef __attribute__((address_space(3))) v4i lv4i;  // LDS    16B payload
#endif

__device__ __forceinline__ void async_copy16(const void* src, void* dst) {
#if HAVE_ASYNC_LDS
    __builtin_amdgcn_global_load_async_to_lds_b128((gv4i*)src, (lv4i*)dst, 0, 0);
#else
    *(v4i*)dst = *(const v4i*)src;
#endif
}

__device__ __forceinline__ void wait_async0() {
#if HAVE_ASYNC_LDS
#if __has_builtin(__builtin_amdgcn_s_wait_asynccnt)
    __builtin_amdgcn_s_wait_asynccnt(0);
#else
    asm volatile("s_wait_asynccnt 0x0" ::: "memory");
#endif
#endif
}

__device__ __forceinline__ float elu_f(float v) {
    return v > 0.f ? v : (expf(v) - 1.f);
}

// ---------------------------------------------------------------------------
// Router: one wave per token; top-2 experts, renormalized weights, compacted
// per-expert token lists (order-independent downstream => deterministic).
// ---------------------------------------------------------------------------
__global__ __launch_bounds__(128)
void router_kernel(const float* __restrict__ x,
                   const float* __restrict__ gate,
                   int*   __restrict__ counts,
                   int*   __restrict__ tok_id,
                   float* __restrict__ tok_wt,
                   int*   __restrict__ tok_slot)
{
    const int t = blockIdx.x * (blockDim.x >> 5) + (threadIdx.x >> 5);
    if (t >= NT) return;
    const int lane = threadIdx.x & 31;

    float s[NE];
#pragma unroll
    for (int j = 0; j < NE; ++j) s[j] = 0.f;

    const float* xr = x + (size_t)t * IN_F;
    for (int d = lane; d < IN_F; d += 32) {
        const float xv = xr[d];
        const float4* g4 = (const float4*)(gate + (size_t)d * NE);
        const float4 g0 = g4[0];
        const float4 g1 = g4[1];
        s[0] += xv * g0.x; s[1] += xv * g0.y; s[2] += xv * g0.z; s[3] += xv * g0.w;
        s[4] += xv * g1.x; s[5] += xv * g1.y; s[6] += xv * g1.z; s[7] += xv * g1.w;
    }
#pragma unroll
    for (int off = 16; off >= 1; off >>= 1) {
#pragma unroll
        for (int j = 0; j < NE; ++j) s[j] += __shfl_xor(s[j], off, 32);
    }

    if (lane == 0) {
        int e0 = 0;
#pragma unroll
        for (int j = 1; j < NE; ++j) if (s[j] > s[e0]) e0 = j;
        int e1 = (e0 == 0) ? 1 : 0;
#pragma unroll
        for (int j = 0; j < NE; ++j) if (j != e0 && s[j] > s[e1]) e1 = j;
        const float w0 = 1.f / (1.f + expf(s[e1] - s[e0]));  // p0/(p0+p1)
        const float w1 = 1.f - w0;

        int p0 = atomicAdd(&counts[e0], 1);
        tok_id[e0 * NT + p0] = t; tok_wt[e0 * NT + p0] = w0; tok_slot[e0 * NT + p0] = 0;
        int p1 = atomicAdd(&counts[e1], 1);
        tok_id[e1 * NT + p1] = t; tok_wt[e1 * NT + p1] = w1; tok_slot[e1 * NT + p1] = 1;
    }
}

// ---------------------------------------------------------------------------
// Expert GEMM, v_wmma_f32_16x16x32_bf16.
//   Block: 256 threads = 8 waves. Tile: BM=256 rows x BN cols, K-step 32.
//   Each wave owns 32 rows (two 16-row A fragments).
//   Double-buffered LDS; loop-invariant row pointers hoisted; async
//   global->LDS (ASYNCcnt) for the bf16 A path.
// ---------------------------------------------------------------------------
template<int BN, bool DO_ELU, bool A_GATHER_F32, bool FINAL>
__global__ __launch_bounds__(256)
void moe_gemm_kernel(const float*  __restrict__ X,        // [NT, K]     (gather)
                     const __bf16* __restrict__ Hsrc,     // [NE*NT, K]  (bf16)
                     const float*  __restrict__ W,        // [NE, K, N]
                     const float*  __restrict__ bias,     // [NE, N]
                     const int*    __restrict__ counts,
                     const int*    __restrict__ tok_id,   // [NE, NT]
                     const float*  __restrict__ tok_wt,
                     const int*    __restrict__ tok_slot,
                     __bf16* __restrict__ Hout,           // [NE*NT, N]  (!FINAL)
                     float*  __restrict__ partial,        // [2, NT, N]  (FINAL)
                     int K, int N)
{
    constexpr int BM = 256;
    constexpr int NB = 32 * BN / 4 / 256;   // float4 staging tasks per thread (B)
    const int e   = blockIdx.z;
    const int cnt = counts[e];
    const int i0  = blockIdx.x * BM;
    if (i0 >= cnt) return;
    const int n0  = blockIdx.y * BN;

    const int tid  = threadIdx.x;
    const int lane = tid & 31;
    const int wave = tid >> 5;
    const int l16  = lane & 15;
    const int hi16 = lane >> 4;

    __shared__ alignas(16) __bf16 As[2][BM][32];
    __shared__ alignas(16) __bf16 Bs[2][BN][32];

    // ---- hoisted, loop-invariant staging pointers --------------------------
    // A: 512 chunk tasks (row, half16); this thread owns tasks tid and tid+256.
    const float*  afp[2];
    const __bf16* abp[2];
    bool          avalid[2];
    int           ahalf[2], arowi[2];
#pragma unroll
    for (int s = 0; s < 2; ++s) {
        const int q = tid + s * 256;
        const int row = q >> 1;
        ahalf[s] = q & 1;
        arowi[s] = row;
        const int i = i0 + row;
        avalid[s] = (i < cnt);
        if (A_GATHER_F32) {
            const int tk = avalid[s] ? tok_id[e * NT + i] : 0;
            afp[s] = X + (size_t)tk * K;
            abp[s] = nullptr;
        } else {
            afp[s] = nullptr;
            abp[s] = Hsrc + ((size_t)e * NT + i) * K;
        }
    }
    // B: this thread owns NB float4 tasks.
    const float* bfp[NB];
    int bkk[NB], bnn[NB];
#pragma unroll
    for (int s = 0; s < NB; ++s) {
        const int pos = (tid + s * 256) * 4;
        bkk[s] = pos / BN;
        bnn[s] = pos % BN;
        bfp[s] = W + ((size_t)e * K + bkk[s]) * N + n0 + bnn[s];
    }

    auto stage = [&](int b, int k0) {
        // ---- A tile: BM x 32 bf16 -----------------------------------------
#pragma unroll
        for (int s = 0; s < 2; ++s) {
            const int kbase = k0 + 16 * ahalf[s];
            __bf16* dst = &As[b][arowi[s]][16 * ahalf[s]];
            const bool ok = avalid[s] && (kbase < K);   // K%16==0: all-or-nothing
            if (ok) {
                if (A_GATHER_F32) {
                    const float* ap = afp[s] + kbase;
                    v8bf r0, r1;
#pragma unroll
                    for (int q = 0; q < 2; ++q) {
                        const float4 f0 = *(const float4*)(ap + 8 * q);
                        const float4 f1 = *(const float4*)(ap + 8 * q + 4);
                        v8bf r;
                        r[0] = (__bf16)f0.x; r[1] = (__bf16)f0.y;
                        r[2] = (__bf16)f0.z; r[3] = (__bf16)f0.w;
                        r[4] = (__bf16)f1.x; r[5] = (__bf16)f1.y;
                        r[6] = (__bf16)f1.z; r[7] = (__bf16)f1.w;
                        if (q == 0) r0 = r; else r1 = r;
                    }
                    *(v8bf*)dst       = r0;
                    *(v8bf*)(dst + 8) = r1;
                } else {
                    const __bf16* sp = abp[s] + kbase;
                    async_copy16(sp, dst);          // 16 bf16 = 2 x b128
                    async_copy16(sp + 8, dst + 8);
                }
            } else {
                v8bf z;
#pragma unroll
                for (int j = 0; j < 8; ++j) z[j] = (__bf16)0.f;
                *(v8bf*)dst       = z;
                *(v8bf*)(dst + 8) = z;
            }
        }
        // ---- B tile: 32 x BN fp32 -> bf16, transposed to [n][k] ------------
#pragma unroll
        for (int s = 0; s < NB; ++s) {
            float4 v = make_float4(0.f, 0.f, 0.f, 0.f);
            if (k0 + bkk[s] < K) v = *(const float4*)(bfp[s] + (size_t)k0 * N);
            const int n = bnn[s], kk = bkk[s];
            Bs[b][n + 0][kk] = (__bf16)v.x;
            Bs[b][n + 1][kk] = (__bf16)v.y;
            Bs[b][n + 2][kk] = (__bf16)v.z;
            Bs[b][n + 3][kk] = (__bf16)v.w;
        }
        // prefetch the weight tile after next (global_prefetch_b8)
        if (k0 + 32 < K)
            __builtin_prefetch(bfp[0] + (size_t)(k0 + 32) * N, 0, 1);
    };

    v8f acc[2][BN / 16] = {};

    stage(0, 0);
    int buf = 0;
    for (int k0 = 0; k0 < K; k0 += 32) {
        if (!A_GATHER_F32) wait_async0();
        __syncthreads();
        if (k0 + 32 < K) stage(buf ^ 1, k0 + 32);

        // A fragments: lanes 0-15 (M=l16) K 0..7 & 16..23; lanes 16-31 K 8..15 & 24..31
        v16bf a[2];
#pragma unroll
        for (int rg = 0; rg < 2; ++rg) {
            const int arow = wave * 32 + rg * 16 + l16;
            const v8bf a0 = *(const v8bf*)&As[buf][arow][8 * hi16];
            const v8bf a1 = *(const v8bf*)&As[buf][arow][16 + 8 * hi16];
            a[rg] = __builtin_shufflevector(a0, a1,
                0, 1, 2, 3, 4, 5, 6, 7, 8, 9, 10, 11, 12, 13, 14, 15);
        }
#pragma unroll
        for (int t = 0; t < BN / 16; ++t) {
            const v8bf b0 = *(const v8bf*)&Bs[buf][t * 16 + l16][8 * hi16];
            const v8bf b1 = *(const v8bf*)&Bs[buf][t * 16 + l16][16 + 8 * hi16];
            const v16bf bb = __builtin_shufflevector(b0, b1,
                0, 1, 2, 3, 4, 5, 6, 7, 8, 9, 10, 11, 12, 13, 14, 15);
#pragma unroll
            for (int rg = 0; rg < 2; ++rg)
                acc[rg][t] = __builtin_amdgcn_wmma_f32_16x16x32_bf16(
                    false, a[rg], false, bb, (short)0, acc[rg][t], false, false);
        }
        buf ^= 1;
    }

    // ---- epilogue: C layout: VGPR r -> (M = r + 8*hi16, N = l16) -----------
    float bv[BN / 16];
#pragma unroll
    for (int t = 0; t < BN / 16; ++t)
        bv[t] = bias[e * N + n0 + t * 16 + l16];

#pragma unroll
    for (int rg = 0; rg < 2; ++rg) {
#pragma unroll
        for (int r = 0; r < 8; ++r) {
            const int m = wave * 32 + rg * 16 + hi16 * 8 + r;
            const int i = i0 + m;
            if (i >= cnt) continue;
            int tok = 0, slot = 0; float wg = 0.f;
            if (FINAL) {
                const int ei = e * NT + i;
                tok = tok_id[ei]; slot = tok_slot[ei]; wg = tok_wt[ei];
            }
#pragma unroll
            for (int t = 0; t < BN / 16; ++t) {
                const int col = n0 + t * 16 + l16;
                float v = acc[rg][t][r] + bv[t];
                if (DO_ELU) v = elu_f(v);
                if (FINAL) {
                    partial[((size_t)slot * NT + tok) * N + col] = v * wg;
                } else {
                    Hout[((size_t)e * NT + i) * N + col] = (__bf16)v;
                }
            }
        }
    }
}

// ---------------------------------------------------------------------------
__global__ void combine_kernel(const float* __restrict__ partial,
                               float* __restrict__ out, int total)
{
    const int i = blockIdx.x * blockDim.x + threadIdx.x;
    if (i < total) out[i] = partial[i] + partial[total + i];
}

// ---------------------------------------------------------------------------
static inline size_t align_up(size_t x, size_t a) { return (x + a - 1) & ~(a - 1); }

extern "C" void kernel_launch(void* const* d_in, const int* in_sizes, int n_in,
                              void* d_out, int out_size, void* d_ws, size_t ws_size,
                              hipStream_t stream)
{
    (void)in_sizes; (void)n_in; (void)out_size; (void)ws_size;

    const float* x      = (const float*)d_in[0];
    const float* gate_w = (const float*)d_in[1];
    const float* w1     = (const float*)d_in[2];
    const float* b1     = (const float*)d_in[3];
    const float* w2     = (const float*)d_in[4];
    const float* b2     = (const float*)d_in[5];
    const float* w3     = (const float*)d_in[6];
    const float* b3     = (const float*)d_in[7];
    const float* w4     = (const float*)d_in[8];
    const float* b4     = (const float*)d_in[9];
    float* out = (float*)d_out;

    char* p = (char*)d_ws;
    size_t off = 0;
    auto carve = [&](size_t bytes) -> char* {
        char* r = p + off;
        off = align_up(off + bytes, 256);
        return r;
    };
    int*    counts   = (int*)   carve(NE * sizeof(int));
    int*    tok_id   = (int*)   carve((size_t)NE * NT * sizeof(int));
    float*  tok_wt   = (float*) carve((size_t)NE * NT * sizeof(float));
    int*    tok_slot = (int*)   carve((size_t)NE * NT * sizeof(int));
    __bf16* H1       = (__bf16*)carve((size_t)NE * NT * INT_F * sizeof(__bf16));
    __bf16* H2       = (__bf16*)carve((size_t)NE * NT * INT_F * sizeof(__bf16));
    float*  partial  = (float*) carve((size_t)2 * NT * OUT_F * sizeof(float));

    (void)hipMemsetAsync(counts, 0, NE * sizeof(int), stream);

    router_kernel<<<NT / 4, 128, 0, stream>>>(x, gate_w, counts, tok_id, tok_wt, tok_slot);

    // Layer 1: [cnt_e, 12336] x [12336, 1024] + b1, ELU (A = gathered fp32 x)
    moe_gemm_kernel<64, true, true, false>
        <<<dim3(NT / 256, INT_F / 64, NE), 256, 0, stream>>>(
            x, nullptr, w1, b1, counts, tok_id, tok_wt, tok_slot,
            H1, nullptr, IN_F, INT_F);

    // Layer 2
    moe_gemm_kernel<64, true, false, false>
        <<<dim3(NT / 256, INT_F / 64, NE), 256, 0, stream>>>(
            nullptr, H1, w2, b2, counts, tok_id, tok_wt, tok_slot,
            H2, nullptr, INT_F, INT_F);

    // Layer 3 (output aliases H1)
    moe_gemm_kernel<64, true, false, false>
        <<<dim3(NT / 256, INT_F / 64, NE), 256, 0, stream>>>(
            nullptr, H2, w3, b3, counts, tok_id, tok_wt, tok_slot,
            H1, nullptr, INT_F, INT_F);

    // Layer 4: + b4, scale by routing weight, scatter to slot-partial buffer
    moe_gemm_kernel<96, false, false, true>
        <<<dim3(NT / 256, 1, NE), 256, 0, stream>>>(
            nullptr, H1, w4, b4, counts, tok_id, tok_wt, tok_slot,
            nullptr, partial, INT_F, OUT_F);

    const int total = NT * OUT_F;
    combine_kernel<<<(total + 255) / 256, 256, 0, stream>>>(partial, out, total);
}
